// TemporalCausalGraph_62740882260118
// MI455X (gfx1250) — matrix operand order repaired and checked by reading, not compiled, not run on hardware
//
#include <hip/hip_runtime.h>
#include <hip/hip_bf16.h>
#include <math.h>

// Problem constants (match reference)
#define Tn   10
#define Sdim 4
#define Bdim 1024
#define Ndim 1024

typedef unsigned short u16;
typedef __attribute__((ext_vector_type(16))) __bf16 v16bf;
typedef __attribute__((ext_vector_type(8)))  float  v8f;

union Frag { uint4 q[2]; v16bf v; };

__device__ __forceinline__ u16 f2bf(float x) {
    unsigned u = __float_as_uint(x);
    u += 0x7FFFu + ((u >> 16) & 1u);          // round-to-nearest-even
    return (u16)(u >> 16);
}
__device__ __forceinline__ float bf2f(u16 h) {
    return __uint_as_float(((unsigned)h) << 16);
}

// ---------------------------------------------------------------------------
// Kernel 1: feats = mean_s X[t,s,b,n]; center over b; split into bf16 hi/lo,
// stored TRANSPOSED as cT[t][n][b]; also ss[t][n] = sum_b c^2.
// ---------------------------------------------------------------------------
__global__ __launch_bounds__(256)
void tcg_prep_kernel(const float* __restrict__ X,
                     u16* __restrict__ cHi, u16* __restrict__ cLo,
                     float* __restrict__ ss) {
    const int t = blockIdx.y;
    const int n = blockIdx.x * 256 + threadIdx.x;
    const size_t SB = (size_t)Bdim * Ndim;
    const float* Xt = X + (size_t)t * Sdim * SB;

    float mu = 0.f;
    for (int b = 0; b < Bdim; ++b) {
        size_t o = (size_t)b * Ndim + n;
        float f = 0.25f * (Xt[o] + Xt[o + SB] + Xt[o + 2 * SB] + Xt[o + 3 * SB]);
        mu += f;
    }
    mu *= (1.0f / (float)Bdim);

    u16* hRow = cHi + ((size_t)t * Ndim + n) * Bdim;
    u16* lRow = cLo + ((size_t)t * Ndim + n) * Bdim;
    float ssa = 0.f;
    for (int b = 0; b < Bdim; ++b) {
        size_t o = (size_t)b * Ndim + n;
        float f = 0.25f * (Xt[o] + Xt[o + SB] + Xt[o + 2 * SB] + Xt[o + 3 * SB]);
        float c = f - mu;
        ssa += c * c;
        u16 h = f2bf(c);
        u16 l = f2bf(c - bf2f(h));
        hRow[b] = h;
        lRow[b] = l;
    }
    ss[(size_t)t * Ndim + n] = ssa;
}

// ---------------------------------------------------------------------------
// Kernel 2: fused Gram (bf16x3 WMMA) + corr + edge-MLP + sigmoid + prior blend
// + time weighting. 128x128 tile per 256-thread block (8 waves, 4x2).
// Double-buffered LDS tiles filled by ASYNC global->LDS copies (ASYNCcnt).
// ---------------------------------------------------------------------------
#define TILE    128
#define KSTEP   32
#define LROW    40          // LDS row stride in bf16 elems (80 B: b128-aligned)
#define TILE_SZ (TILE * LROW)              // u16 elems per tile buffer
// dynamic smem: 2 bufs x 4 mats x TILE_SZ u16  +  float region
#define SMEM_BYTES (2 * 4 * TILE_SZ * 2 + (TILE + TILE + 32 + 16 + 16 + 4) * 4)

__global__ __launch_bounds__(256)
void tcg_gram_fused_kernel(const u16* __restrict__ cHi, const u16* __restrict__ cLo,
                           const float* __restrict__ ss,
                           const float* __restrict__ eNow, const float* __restrict__ eLag,
                           const float* __restrict__ prior,
                           const float* __restrict__ w1, const float* __restrict__ b1,
                           const float* __restrict__ w2, const float* __restrict__ b2,
                           float* __restrict__ outNow, float* __restrict__ outLag) {
    extern __shared__ __align__(16) u16 smem[];
    // tile buffers: [buf(2)][mat(4)][TILE_SZ]; mats: 0=Ahi 1=Alo 2=Bhi 3=Blo
    u16* tiles = smem;
    float* sSSI = (float*)(smem + 8 * TILE_SZ);
    float* sSSJ = sSSI + TILE;
    float* sW1  = sSSJ + TILE;
    float* sB1  = sW1 + 32;
    float* sW2  = sB1 + 16;
    float* sB2  = sW2 + 16;

    const int tid  = threadIdx.x;
    const int lane = tid & 31;
    const int wave = tid >> 5;
    const int wi   = wave >> 1;      // 0..3  (32-row band)
    const int wj   = wave & 1;       // 0..1  (64-col band)
    const int i0   = blockIdx.y * TILE;
    const int j0   = blockIdx.x * TILE;
    const int m    = lane & 15;
    const int half = lane >> 4;

    if (tid < 32) sW1[tid] = w1[tid];
    if (tid < 16) { sB1[tid] = b1[tid]; sW2[tid] = w2[tid]; }
    if (tid == 0) sB2[0] = b2[0];

    // per-thread staging geometry: 2 chunks of 16 B per matrix per k-step
    int rowc[2], chc[2];
#pragma unroll
    for (int r = 0; r < 2; ++r) {
        int cidx = tid + 256 * r;          // 0..511
        rowc[r] = cidx >> 2;               // 0..127
        chc[r]  = (cidx & 3) * 8;          // bf16 offset within 32-wide row
    }

    float lagAcc[2][4][8];
#pragma unroll
    for (int si = 0; si < 2; ++si)
#pragma unroll
        for (int sj = 0; sj < 4; ++sj)
#pragma unroll
            for (int r = 0; r < 8; ++r) lagAcc[si][sj][r] = 0.f;

    const v8f vzero = {0.f, 0.f, 0.f, 0.f, 0.f, 0.f, 0.f, 0.f};

    for (int t = 0; t < Tn; ++t) {
        __syncthreads();   // previous t's tile/ss readers are done

        // global row-base pointers for this t
        const u16* gA_hi = cHi + ((size_t)t * Ndim + i0) * Bdim;
        const u16* gA_lo = cLo + ((size_t)t * Ndim + i0) * Bdim;
        const u16* gB_hi = cHi + ((size_t)t * Ndim + j0) * Bdim;
        const u16* gB_lo = cLo + ((size_t)t * Ndim + j0) * Bdim;

        // async-stage one 128xKSTEP tile set into buffer `buf` at column k0
        auto stage = [&](int buf, int k0) {
            const u16* gsrc[4] = {gA_hi, gA_lo, gB_hi, gB_lo};
#pragma unroll
            for (int r = 0; r < 2; ++r) {
                size_t go = (size_t)rowc[r] * Bdim + k0 + chc[r];
                unsigned lbase = (unsigned)(uintptr_t)
                    (tiles + (size_t)buf * 4 * TILE_SZ + rowc[r] * LROW + chc[r]);
#pragma unroll
                for (int mat = 0; mat < 4; ++mat) {
                    unsigned long long ga = (unsigned long long)(uintptr_t)(gsrc[mat] + go);
                    unsigned la = lbase + (unsigned)(mat * TILE_SZ * 2);
                    asm volatile("global_load_async_to_lds_b128 %0, %1, off"
                                 :: "v"(la), "v"(ga) : "memory");
                }
            }
        };

        stage(0, 0);                       // prologue fill of buffer 0 (8 async ops)

        if (tid < TILE) {
            sSSI[tid] = ss[(size_t)t * Ndim + i0 + tid];
            sSSJ[tid] = ss[(size_t)t * Ndim + j0 + tid];
        }

        v8f acc[2][4];
#pragma unroll
        for (int si = 0; si < 2; ++si)
#pragma unroll
            for (int sj = 0; sj < 4; ++sj) acc[si][sj] = vzero;

        for (int k0 = 0; k0 < Bdim; k0 += KSTEP) {
            const int cur = (k0 >> 5) & 1;
            if (k0 + KSTEP < Bdim) {
                stage(cur ^ 1, k0 + KSTEP);               // overlap next tile copy
                asm volatile("s_wait_asynccnt 8" ::: "memory");  // oldest 8 (= cur) done
            } else {
                asm volatile("s_wait_asynccnt 0" ::: "memory");
            }
            __syncthreads();   // all waves' async writes for `cur` visible

            const u16* tAhi = tiles + (size_t)cur * 4 * TILE_SZ + 0 * TILE_SZ;
            const u16* tAlo = tAhi + TILE_SZ;
            const u16* tBhi = tAlo + TILE_SZ;
            const u16* tBlo = tBhi + TILE_SZ;

            Frag aHi[2], aLo[2];
#pragma unroll
            for (int si = 0; si < 2; ++si) {
                int base = (wi * 32 + si * 16 + m) * LROW + half * 8;
                aHi[si].q[0] = *(const uint4*)&tAhi[base];
                aHi[si].q[1] = *(const uint4*)&tAhi[base + 16];
                aLo[si].q[0] = *(const uint4*)&tAlo[base];
                aLo[si].q[1] = *(const uint4*)&tAlo[base + 16];
            }
#pragma unroll
            for (int sj = 0; sj < 4; ++sj) {
                int base = (wj * 64 + sj * 16 + m) * LROW + half * 8;
                Frag bHi, bLo;
                bHi.q[0] = *(const uint4*)&tBhi[base];
                bHi.q[1] = *(const uint4*)&tBhi[base + 16];
                bLo.q[0] = *(const uint4*)&tBlo[base];
                bLo.q[1] = *(const uint4*)&tBlo[base + 16];
#pragma unroll
                for (int si = 0; si < 2; ++si) {
                    // bf16x3: hi*hi + hi*lo + lo*hi  (drop lo*lo)
                    acc[si][sj] = __builtin_amdgcn_wmma_f32_16x16x32_bf16(
                        false, aLo[si].v, false, bHi.v, (short)0, acc[si][sj], false, false);
                    acc[si][sj] = __builtin_amdgcn_wmma_f32_16x16x32_bf16(
                        false, aHi[si].v, false, bLo.v, (short)0, acc[si][sj], false, false);
                    acc[si][sj] = __builtin_amdgcn_wmma_f32_16x16x32_bf16(
                        false, aHi[si].v, false, bHi.v, (short)0, acc[si][sj], false, false);
                }
            }
            __syncthreads();   // readers done before next iter's async overwrites `cur^1`... (next stages into `cur`)
        }

        // ---- fused epilogue for this t ----
        const float wt = 1.0f - 0.1f * (float)t;
        const float* eParam = (t == 0) ? eNow : eLag;
#pragma unroll
        for (int si = 0; si < 2; ++si) {
#pragma unroll
            for (int sj = 0; sj < 4; ++sj) {
                int lj = wj * 64 + sj * 16 + m;
                int gj = j0 + lj;
                float ssj = sSSJ[lj];
#pragma unroll
                for (int r = 0; r < 8; ++r) {
                    int li = wi * 32 + si * 16 + half * 8 + r;   // C layout: VGPR r -> M=r(+8)
                    int gi = i0 + li;
                    float numer = acc[si][sj][r];
                    float corr = fabsf(numer) * rsqrtf(sSSI[li] * ssj + 1e-8f);
                    if (gi == gj) corr = 0.f;
                    corr = fminf(corr, 1.0f);
                    float param = eParam[(size_t)gi * Ndim + gj];
                    // Linear(2,16) -> LeakyReLU(0.01) -> Linear(16,1) -> Sigmoid
                    float z = sB2[0];
                    for (int kk = 0; kk < 16; ++kk) {
                        float h = sW1[2 * kk] * corr + sW1[2 * kk + 1] * param + sB1[kk];
                        h = (h > 0.f) ? h : 0.01f * h;
                        z += sW2[kk] * h;
                    }
                    float s = 1.0f / (1.0f + __expf(-z));
                    if (t == 0) {
                        float sp = 1.0f / (1.0f + __expf(-prior[(size_t)gi * Ndim + gj]));
                        outNow[(size_t)gi * Ndim + gj] =
                            (gi == gj) ? 0.f : (0.7f * s + 0.3f * sp);   // w0 = 1.0
                    } else {
                        lagAcc[si][sj][r] += wt * s;
                    }
                }
            }
        }
    }

    // adj_lag = offdiag * (0.7/9 * sum_{t>=1} w_t s_t + 0.3 * 0.5 * sigmoid(prior))
#pragma unroll
    for (int si = 0; si < 2; ++si) {
#pragma unroll
        for (int sj = 0; sj < 4; ++sj) {
            int gj = j0 + wj * 64 + sj * 16 + m;
#pragma unroll
            for (int r = 0; r < 8; ++r) {
                int gi = i0 + wi * 32 + si * 16 + half * 8 + r;
                float sp = 1.0f / (1.0f + __expf(-prior[(size_t)gi * Ndim + gj]));
                float v = 0.7f * (lagAcc[si][sj][r] * (1.0f / 9.0f)) + 0.15f * sp;
                outLag[(size_t)gi * Ndim + gj] = (gi == gj) ? 0.f : v;
            }
        }
    }
}

// ---------------------------------------------------------------------------
extern "C" void kernel_launch(void* const* d_in, const int* in_sizes, int n_in,
                              void* d_out, int out_size, void* d_ws, size_t ws_size,
                              hipStream_t stream) {
    (void)in_sizes; (void)n_in; (void)out_size; (void)ws_size;

    const float* X     = (const float*)d_in[0];
    // d_in[1] = time_context (unused by the reference)
    const float* eNow  = (const float*)d_in[2];
    const float* eLag  = (const float*)d_in[3];
    const float* prior = (const float*)d_in[4];
    const float* w1    = (const float*)d_in[5];
    const float* b1    = (const float*)d_in[6];
    const float* w2    = (const float*)d_in[7];
    const float* b2    = (const float*)d_in[8];

    // workspace layout: cHi (bf16, T*N*B) | cLo (bf16, T*N*B) | ss (f32, T*N)
    u16*   cHi = (u16*)d_ws;
    u16*   cLo = cHi + (size_t)Tn * Ndim * Bdim;
    float* ss  = (float*)(cLo + (size_t)Tn * Ndim * Bdim);

    float* outNow = (float*)d_out;
    float* outLag = outNow + (size_t)Ndim * Ndim;

    dim3 g1(Ndim / 256, Tn);
    tcg_prep_kernel<<<g1, 256, 0, stream>>>(X, cHi, cLo, ss);

    dim3 g2(Ndim / TILE, Ndim / TILE);   // 8 x 8 tiles of 128x128
    tcg_gram_fused_kernel<<<g2, 256, SMEM_BYTES, stream>>>(cHi, cLo, ss, eNow, eLag,
                                                           prior, w1, b1, w2, b2,
                                                           outNow, outLag);
}
